// Text2SemanticDecoder_58351425683657
// MI455X (gfx1250) — matrix operand order repaired and checked by reading, not compiled, not run on hardware
//
#include <hip/hip_runtime.h>

#define DMODEL 1024
#define DFFN   4096
#define NHEAD  16
#define DHEAD  64
#define SEQ    1024
#define BATCH  4
#define NLAYER 16
#define MROWS  (BATCH * SEQ)   /* 4096 */
#define D3     (3 * DMODEL)    /* 3072 */

typedef __attribute__((ext_vector_type(16))) __bf16 v16bf;
typedef __attribute__((ext_vector_type(8)))  __bf16 v8bf;
typedef __attribute__((ext_vector_type(4)))  __bf16 v4bf;
typedef __attribute__((ext_vector_type(8)))  float  v8f;
typedef __attribute__((ext_vector_type(4)))  float  v4f;

union ABu { v16bf v; v8bf h[2]; };

__device__ __forceinline__ v8f wmma_bf16(v16bf a, v16bf b, v8f c) {
  // D = A(16x32 bf16) x B(32x16 bf16) + C(16x16 f32)
  return __builtin_amdgcn_wmma_f32_16x16x32_bf16(false, a, false, b, (short)0, c,
                                                 false, false);
}

__device__ __forceinline__ v8f zero8() {
  v8f z;
#pragma unroll
  for (int i = 0; i < 8; ++i) z[i] = 0.0f;
  return z;
}

// 16-lane butterfly reductions via DPP (pure VALU, no LDS pipe):
// quad_perm[1,0,3,2] (xor1), quad_perm[2,3,0,1] (xor2), row_half_mirror (8),
// row_mirror (16). All lanes active in callers.
__device__ __forceinline__ float dpp_max16(float x) {
  int t;
  t = __builtin_amdgcn_update_dpp(0, __float_as_int(x), 0xB1, 0xF, 0xF, true);
  x = fmaxf(x, __int_as_float(t));
  t = __builtin_amdgcn_update_dpp(0, __float_as_int(x), 0x4E, 0xF, 0xF, true);
  x = fmaxf(x, __int_as_float(t));
  t = __builtin_amdgcn_update_dpp(0, __float_as_int(x), 0x141, 0xF, 0xF, true);
  x = fmaxf(x, __int_as_float(t));
  t = __builtin_amdgcn_update_dpp(0, __float_as_int(x), 0x140, 0xF, 0xF, true);
  x = fmaxf(x, __int_as_float(t));
  return x;
}

__device__ __forceinline__ float dpp_sum16(float x) {
  int t;
  t = __builtin_amdgcn_update_dpp(0, __float_as_int(x), 0xB1, 0xF, 0xF, true);
  x += __int_as_float(t);
  t = __builtin_amdgcn_update_dpp(0, __float_as_int(x), 0x4E, 0xF, 0xF, true);
  x += __int_as_float(t);
  t = __builtin_amdgcn_update_dpp(0, __float_as_int(x), 0x141, 0xF, 0xF, true);
  x += __int_as_float(t);
  t = __builtin_amdgcn_update_dpp(0, __float_as_int(x), 0x140, 0xF, 0xF, true);
  x += __int_as_float(t);
  return x;
}

// A fragment 16x32 (rows m0..m0+15 of row-major bf16 matrix, K-chunk k0..k0+31).
// Contiguous-K layout used here (A and B match): lane = row (mod 16), lane-half selects K 0..15 / 16..31.
__device__ __forceinline__ v16bf frag_a(const __bf16* __restrict__ A, int lda,
                                        int m0, int k0, int lane) {
  const int r  = m0 + (lane & 15);
  const int k8 = (lane >> 4) * 16;
  const __bf16* p = A + (size_t)r * lda + k0 + k8;
  ABu u;
  u.h[0] = *(const v8bf*)(p);
  u.h[1] = *(const v8bf*)(p + 8);
  return u.v;
}

// B fragment 32x16 taken from W[N,K] row-major (B[k][n] = W[n][k]).
// Contiguous-K layout used here (matches frag_a): lane = column (mod 16) = W row; lane-half selects K 0..15 / 16..31.
__device__ __forceinline__ v16bf frag_b(const __bf16* __restrict__ W, int ldw,
                                        int n0, int k0, int lane) {
  const int n   = n0 + (lane & 15);
  const int k16 = (lane >> 4) * 16;
  const __bf16* p = W + (size_t)n * ldw + k0 + k16;
  ABu u;
  u.h[0] = *(const v8bf*)(p);
  u.h[1] = *(const v8bf*)(p + 8);
  return u.v;
}

// ---------------------------------------------------------------------------
// GEMM: out[M,N] = A[M,K] @ W[N,K]^T (+bias, epilogue variants)
// EPI 0: +bias            -> bf16 out      (QKV)
// EPI 1: +bias +residual  -> f32 out       (out-proj / MLP2)
// EPI 2: +bias, ReLU      -> bf16 out      (MLP1)
// Block = 256 threads (8 waves as 2Mx4N), block tile 128(M) x 256(N);
// wave tile 64x64: 16 WMMA per 16 b128 loads, 128 accumulator VGPRs.
// ---------------------------------------------------------------------------
template <int EPI>
__global__ __launch_bounds__(256) void k_gemm(
    const __bf16* __restrict__ A, const __bf16* __restrict__ W,
    const float* __restrict__ bias, const float* __restrict__ res,
    float* __restrict__ outf, __bf16* __restrict__ outb, int N, int K) {
  const int lane = threadIdx.x & 31;
  const int wave = threadIdx.x >> 5;
  const int m0 = blockIdx.y * 128 + (wave & 1) * 64;
  const int n0 = blockIdx.x * 256 + (wave >> 1) * 64;

  v8f acc[4][4];
#pragma unroll
  for (int i = 0; i < 4; ++i)
#pragma unroll
    for (int j = 0; j < 4; ++j) acc[i][j] = zero8();

  for (int k = 0; k < K; k += 32) {
    if (k + 32 < K) {
      __builtin_prefetch(A + (size_t)(m0 + (lane & 15)) * K + k + 32, 0, 1);
      __builtin_prefetch(W + (size_t)(n0 + (lane & 15)) * K + k + 32, 0, 1);
    }
    v16bf a[4];
#pragma unroll
    for (int i = 0; i < 4; ++i) a[i] = frag_a(A, K, m0 + 16 * i, k, lane);
#pragma unroll
    for (int j = 0; j < 4; ++j) {
      const v16bf b = frag_b(W, K, n0 + j * 16, k, lane);
#pragma unroll
      for (int i = 0; i < 4; ++i) acc[i][j] = wmma_bf16(a[i], b, acc[i][j]);
    }
  }

  const int hl = lane >> 4;
  const int nl = lane & 15;
#pragma unroll
  for (int i = 0; i < 4; ++i) {
#pragma unroll
    for (int j = 0; j < 4; ++j) {
      const int n = n0 + j * 16 + nl;
      const float bv = bias[n];
#pragma unroll
      for (int v = 0; v < 8; ++v) {
        const int m = m0 + i * 16 + v + 8 * hl;  // C layout: M = vgpr + 8*half
        float val = acc[i][j][v] + bv;
        if (EPI == 1) val += res[(size_t)m * N + n];
        if (EPI == 2) val = fmaxf(val, 0.0f);
        if (EPI == 1) outf[(size_t)m * N + n] = val;
        else          outb[(size_t)m * N + n] = (__bf16)val;
      }
    }
  }
}

// ---------------------------------------------------------------------------
// Fused causal flash attention. Grid: (B*H, SEQ/64), block 128 = 4 waves,
// each wave owns one 16-query tile; 64 keys per iteration.
// qkv: [B*S, 3D] bf16; vt: [B*H*dh, S] bf16.
// - mask-free bulk + single diagonal tail block
// - row max via DPP butterflies (no ds_bpermute)
// - row sums via P x ones WMMA (result lands in the same C layout as rl)
// ---------------------------------------------------------------------------
__global__ __launch_bounds__(128) void k_attn(const __bf16* __restrict__ qkv,
                                              const __bf16* __restrict__ vt,
                                              __bf16* __restrict__ ctx) {
  __shared__ __align__(16) __bf16 Plds[4][16][64];
  const int lane = threadIdx.x & 31;
  const int wave = threadIdx.x >> 5;
  const int bh = blockIdx.x;  // 0..63
  const int b = bh >> 4, hh = bh & 15;
  const int q0 = (blockIdx.y * 4 + wave) * 16;
  const int r = lane & 15;
  const int k8 = (lane >> 4) * 16;

  // Q fragments, pre-scaled by 1/sqrt(dh)=0.125 (exact in bf16)
  const __bf16* qrow = qkv + (size_t)(b * SEQ + q0 + r) * D3 + hh * DHEAD + k8;
  v16bf qf[2];
#pragma unroll
  for (int f = 0; f < 2; ++f) {
    ABu u;
    u.h[0] = *(const v8bf*)(qrow + f * 32);
    u.h[1] = *(const v8bf*)(qrow + f * 32 + 8);
#pragma unroll
    for (int e = 0; e < 16; ++e) u.v[e] = (__bf16)((float)u.v[e] * 0.125f);
    qf[f] = u.v;
  }
  // all-ones B fragment for row-sum WMMA
  ABu ou;
#pragma unroll
  for (int e = 0; e < 16; ++e) ou.v[e] = (__bf16)1.0f;
  const v16bf onesb = ou.v;

  const __bf16* kbase = qkv + (size_t)(b * SEQ) * D3 + DMODEL + hh * DHEAD;
  const __bf16* vbase = vt + (size_t)bh * DHEAD * SEQ;

  float rm[8], rl[8];
  v8f cacc[4];
#pragma unroll
  for (int v = 0; v < 8; ++v) { rm[v] = -1e30f; rl[v] = 0.0f; }
#pragma unroll
  for (int d = 0; d < 4; ++d) cacc[d] = zero8();

  auto process = [&](int k0, bool DOMASK) {
    // scores S[16,64] = (Q*scale) @ K^T : four 16x16 tiles, 2 WMMAs each
    v8f s[4];
#pragma unroll
    for (int t = 0; t < 4; ++t) {
      const __bf16* kp =
          kbase + (size_t)(k0 + t * 16 + (lane & 15)) * D3 + ((lane >> 4) << 4);
      ABu u;
      u.h[0] = *(const v8bf*)(kp);
      u.h[1] = *(const v8bf*)(kp + 8);
      s[t] = wmma_bf16(qf[0], u.v, zero8());
      u.h[0] = *(const v8bf*)(kp + 32);
      u.h[1] = *(const v8bf*)(kp + 40);
      s[t] = wmma_bf16(qf[1], u.v, s[t]);
    }
    // online softmax; rows live in (vgpr index + 8*lane-half)
    float af[8];
#pragma unroll
    for (int v = 0; v < 8; ++v) {
      float e[4];
#pragma unroll
      for (int t = 0; t < 4; ++t) {
        e[t] = s[t][v];
        if (DOMASK) {
          const int qr = q0 + v + 8 * (lane >> 4);
          if (k0 + t * 16 + (lane & 15) > qr) e[t] = -1e30f;
        }
      }
      float mx = fmaxf(fmaxf(e[0], e[1]), fmaxf(e[2], e[3]));
      mx = dpp_max16(mx);
      const float mnew = fmaxf(rm[v], mx);
      af[v] = __expf(rm[v] - mnew);
      rm[v] = mnew;
#pragma unroll
      for (int t = 0; t < 4; ++t) s[t][v] = __expf(e[t] - mnew);
    }
#pragma unroll
    for (int d = 0; d < 4; ++d)
#pragma unroll
      for (int v = 0; v < 8; ++v) cacc[d][v] *= af[v];

    // re-layout P: C-fragment -> LDS -> two A-fragments (same wave; DS in-order)
#pragma unroll
    for (int t = 0; t < 4; ++t)
#pragma unroll
      for (int v = 0; v < 8; ++v)
        Plds[wave][v + 8 * (lane >> 4)][t * 16 + (lane & 15)] = (__bf16)s[t][v];
    ABu pu0, pu1;
    pu0.h[0] = *(const v8bf*)&Plds[wave][r][k8];
    pu0.h[1] = *(const v8bf*)&Plds[wave][r][k8 + 8];
    pu1.h[0] = *(const v8bf*)&Plds[wave][r][32 + k8];
    pu1.h[1] = *(const v8bf*)&Plds[wave][r][32 + k8 + 8];

    // row sums via WMMA: (P @ ones)[m][*] = sum_k P[m][k], same layout as rl
    v8f sums = wmma_bf16(pu0.v, onesb, zero8());
    sums = wmma_bf16(pu1.v, onesb, sums);
#pragma unroll
    for (int v = 0; v < 8; ++v) rl[v] = rl[v] * af[v] + sums[v];

    // ctx += P(16x64) @ V(64x64): B-frags from vt rows (dh index), contig in key
#pragma unroll
    for (int d = 0; d < 4; ++d) {
      const __bf16* vp =
          vbase + (size_t)(d * 16 + (lane & 15)) * SEQ + k0 + ((lane >> 4) << 4);
      ABu u;
      u.h[0] = *(const v8bf*)(vp);
      u.h[1] = *(const v8bf*)(vp + 8);
      cacc[d] = wmma_bf16(pu0.v, u.v, cacc[d]);
      u.h[0] = *(const v8bf*)(vp + 32);
      u.h[1] = *(const v8bf*)(vp + 40);
      cacc[d] = wmma_bf16(pu1.v, u.v, cacc[d]);
    }
  };

  const int kb_full = q0 >> 6;  // fully-unmasked 64-key blocks
  for (int kb = 0; kb < kb_full; ++kb) process(kb << 6, false);
  process(kb_full << 6, true);  // single diagonal tail block

  // normalize and scatter to ctx[B*S, D]
#pragma unroll
  for (int v = 0; v < 8; ++v) {
    const float inv = 1.0f / rl[v];
    const int m = q0 + v + 8 * (lane >> 4);
#pragma unroll
    for (int d = 0; d < 4; ++d) {
      const int col = hh * DHEAD + d * 16 + (lane & 15);
      ctx[(size_t)(b * SEQ + m) * DMODEL + col] = (__bf16)(cacc[d][v] * inv);
    }
  }
}

// V slice of qkv -> vt[B,H,dh,S] (transposed for contiguous-K B fragments)
__global__ __launch_bounds__(256) void k_vt(const __bf16* __restrict__ qkv,
                                            __bf16* __restrict__ vt) {
  const int idx = blockIdx.x * 256 + threadIdx.x;  // B*H*dh*S = 4M
  const int s = idx & (SEQ - 1);
  const int d = (idx >> 10) & (DHEAD - 1);
  const int bh = idx >> 16;
  const int b = bh >> 4, hh = bh & 15;
  vt[idx] = qkv[(size_t)(b * SEQ + s) * D3 + 2 * DMODEL + hh * DHEAD + d];
}

// f32 -> bf16, 4 elements/thread (b128 load, b64 store)
__global__ __launch_bounds__(256) void k_cvt(const float* __restrict__ s,
                                             __bf16* __restrict__ d, int n4) {
  const int i = blockIdx.x * 256 + threadIdx.x;
  if (i < n4) {
    const v4f x = ((const v4f*)s)[i];
    v4bf y;
#pragma unroll
    for (int e = 0; e < 4; ++e) y[e] = (__bf16)x[e];
    ((v4bf*)d)[i] = y;
  }
}

__global__ __launch_bounds__(256) void k_init(const float* __restrict__ x,
                                              float* __restrict__ h,
                                              __bf16* __restrict__ hb, int n4) {
  const int i = blockIdx.x * 256 + threadIdx.x;
  if (i < n4) {
    const v4f v = ((const v4f*)x)[i];
    v4bf y;
#pragma unroll
    for (int e = 0; e < 4; ++e) y[e] = (__bf16)v[e];
    ((v4f*)h)[i] = v;
    ((v4bf*)hb)[i] = y;
  }
}

// LayerNorm over D=1024, one row per block of 256 threads (4 elems/thread,
// vector loads/stores); DPP 16-lane reduce + 16-partial shared combine.
__global__ __launch_bounds__(256) void k_ln(const float* __restrict__ in,
                                            const float* __restrict__ w,
                                            const float* __restrict__ bias,
                                            float* __restrict__ outf,
                                            __bf16* __restrict__ outb) {
  __shared__ float shm[32];
  const int row = blockIdx.x;
  const float* x = in + (size_t)row * DMODEL;
  const v4f xv = ((const v4f*)x)[threadIdx.x];
  float s = xv[0] + xv[1] + xv[2] + xv[3];
  float s2 = xv[0] * xv[0] + xv[1] * xv[1] + xv[2] * xv[2] + xv[3] * xv[3];
  s = dpp_sum16(s);
  s2 = dpp_sum16(s2);
  if ((threadIdx.x & 15) == 0) {
    shm[threadIdx.x >> 4] = s;
    shm[16 + (threadIdx.x >> 4)] = s2;
  }
  __syncthreads();
  float ts = 0.0f, ts2 = 0.0f;
#pragma unroll
  for (int i = 0; i < 16; ++i) {
    ts += shm[i];
    ts2 += shm[16 + i];
  }
  const float mean = ts * (1.0f / DMODEL);
  const float var = ts2 * (1.0f / DMODEL) - mean * mean;
  const float inv = rsqrtf(var + 1e-5f);
  const v4f wv = ((const v4f*)w)[threadIdx.x];
  const v4f bv = ((const v4f*)bias)[threadIdx.x];
  v4f y;
  v4bf yb;
#pragma unroll
  for (int e = 0; e < 4; ++e) {
    y[e] = (xv[e] - mean) * inv * wv[e] + bv[e];
    yb[e] = (__bf16)y[e];
  }
  ((v4f*)(outf + (size_t)row * DMODEL))[threadIdx.x] = y;
  ((v4bf*)(outb + (size_t)row * DMODEL))[threadIdx.x] = yb;
}

extern "C" void kernel_launch(void* const* d_in, const int* in_sizes, int n_in,
                              void* d_out, int out_size, void* d_ws,
                              size_t ws_size, hipStream_t stream) {
  (void)in_sizes; (void)n_in; (void)out_size; (void)ws_size;
  const float* x     = (const float*)d_in[0];
  const float* qkv_w = (const float*)d_in[1];
  const float* qkv_b = (const float*)d_in[2];
  const float* out_w = (const float*)d_in[3];
  const float* out_b = (const float*)d_in[4];
  const float* w1    = (const float*)d_in[5];
  const float* b1    = (const float*)d_in[6];
  const float* w2    = (const float*)d_in[7];
  const float* b2    = (const float*)d_in[8];
  const float* ln1w  = (const float*)d_in[9];
  const float* ln1b  = (const float*)d_in[10];
  const float* ln2w  = (const float*)d_in[11];
  const float* ln2b  = (const float*)d_in[12];

  char* p = (char*)d_ws;
  auto alloc = [&](size_t bytes) {
    char* q = p;
    p += (bytes + 255) & ~(size_t)255;
    return q;
  };
  float*  h    = (float*) alloc((size_t)MROWS * DMODEL * 4);
  float*  h2   = (float*) alloc((size_t)MROWS * DMODEL * 4);
  __bf16* hb   = (__bf16*)alloc((size_t)MROWS * DMODEL * 2);
  __bf16* qkvb = (__bf16*)alloc((size_t)MROWS * D3 * 2);
  __bf16* vtb  = (__bf16*)alloc((size_t)BATCH * NHEAD * DHEAD * SEQ * 2);
  __bf16* ctxb = (__bf16*)alloc((size_t)MROWS * DMODEL * 2);
  __bf16* mlpb = (__bf16*)alloc((size_t)MROWS * DFFN * 2);
  __bf16* wqkv = (__bf16*)alloc((size_t)D3 * DMODEL * 2);
  __bf16* wout = (__bf16*)alloc((size_t)DMODEL * DMODEL * 2);
  __bf16* w1b  = (__bf16*)alloc((size_t)DFFN * DMODEL * 2);
  __bf16* w2b  = (__bf16*)alloc((size_t)DMODEL * DFFN * 2);

  k_init<<<MROWS * DMODEL / 1024, 256, 0, stream>>>(x, h, hb,
                                                    MROWS * DMODEL / 4);

  for (int l = 0; l < NLAYER; ++l) {
    const float* Wq = qkv_w + (size_t)l * D3 * DMODEL;
    const float* Wo = out_w + (size_t)l * DMODEL * DMODEL;
    const float* W1 = w1 + (size_t)l * DFFN * DMODEL;
    const float* W2 = w2 + (size_t)l * DMODEL * DFFN;

    k_cvt<<<D3 * DMODEL / 1024, 256, 0, stream>>>(Wq, wqkv, D3 * DMODEL / 4);
    k_cvt<<<DMODEL * DMODEL / 1024, 256, 0, stream>>>(Wo, wout,
                                                      DMODEL * DMODEL / 4);
    k_cvt<<<DFFN * DMODEL / 1024, 256, 0, stream>>>(W1, w1b, DFFN * DMODEL / 4);
    k_cvt<<<DMODEL * DFFN / 1024, 256, 0, stream>>>(W2, w2b, DMODEL * DFFN / 4);

    // qkv = h @ Wq^T + bq   (bf16 out)
    k_gemm<0><<<dim3(D3 / 256, MROWS / 128), 256, 0, stream>>>(
        hb, wqkv, qkv_b + (size_t)l * D3, nullptr, nullptr, qkvb, D3, DMODEL);
    k_vt<<<BATCH * NHEAD * DHEAD * SEQ / 256, 256, 0, stream>>>(qkvb, vtb);
    k_attn<<<dim3(BATCH * NHEAD, SEQ / 64), 128, 0, stream>>>(qkvb, vtb, ctxb);
    // h2 = h + ctx @ Wo^T + bo   (f32 out)
    k_gemm<1><<<dim3(DMODEL / 256, MROWS / 128), 256, 0, stream>>>(
        ctxb, wout, out_b + (size_t)l * DMODEL, h, h2, nullptr, DMODEL, DMODEL);
    k_ln<<<MROWS, 256, 0, stream>>>(h2, ln1w + (size_t)l * DMODEL,
                                    ln1b + (size_t)l * DMODEL, h, hb);
    // mlp = relu(h @ W1^T + b1)  (bf16 out)
    k_gemm<2><<<dim3(DFFN / 256, MROWS / 128), 256, 0, stream>>>(
        hb, w1b, b1 + (size_t)l * DFFN, nullptr, nullptr, mlpb, DFFN, DMODEL);
    // h2 = h + mlp @ W2^T + b2   (f32 out)
    k_gemm<1><<<dim3(DMODEL / 256, MROWS / 128), 256, 0, stream>>>(
        mlpb, w2b, b2 + (size_t)l * DMODEL, h, h2, nullptr, DMODEL, DFFN);
    float* lnout = (l == NLAYER - 1) ? (float*)d_out : h;
    k_ln<<<MROWS, 256, 0, stream>>>(h2, ln2w + (size_t)l * DMODEL,
                                    ln2b + (size_t)l * DMODEL, lnout, hb);
  }
}
